// RECENTModel_16234976379135
// MI455X (gfx1250) — matrix operand order repaired and checked by reading, not compiled, not run on hardware
//
#include <hip/hip_runtime.h>

typedef __attribute__((ext_vector_type(16))) _Float16 v16h;
typedef __attribute__((ext_vector_type(8)))  _Float16 v8h;
typedef __attribute__((ext_vector_type(8)))  float    v8f;
typedef __attribute__((ext_vector_type(4)))  float    v4f;

#define BATCH 65536
#define HID   1024
#define NLAB  30
#define MAXO  19
#define NH    12
#define NCOL  138    // sum of per-head widths
#define NCOLP 144    // padded to 9 N-tiles of 16
#define NT    9
#define KG    128    // K-slice staged in LDS per barrier round
#define SBS   136    // LDS column stride in halves (16B aligned, conflict-free)

__constant__ int c_n[NH]   = {9,9,7,8,9,8,16,16,13,19,15,9};
__constant__ int c_off[NH] = {0,9,18,25,33,42,50,66,82,95,114,129};

__constant__ int c_cand[NH][MAXO] = {
  {0,1,2,3,5,7,19,20,28, 0,0,0,0,0,0,0,0,0,0},
  {0,1,2,3,5,7,19,20,28, 0,0,0,0,0,0,0,0,0,0},
  {0,5,7,18,19,20,22, 0,0,0,0,0,0,0,0,0,0,0,0},
  {0,1,2,3,5,7,19,20, 0,0,0,0,0,0,0,0,0,0,0},
  {0,1,2,3,5,7,19,20,28, 0,0,0,0,0,0,0,0,0,0},
  {0,1,2,3,5,7,9,20, 0,0,0,0,0,0,0,0,0,0,0},
  {0,2,4,6,8,10,12,13,14,15,16,17,21,24,26,27, 0,0,0},
  {0,4,6,8,10,11,12,13,14,15,17,23,24,26,27,29, 0,0,0},
  {0,4,6,10,11,14,15,17,21,24,25,26,27, 0,0,0,0,0,0},
  {0,4,6,7,8,10,11,12,13,14,15,17,21,23,24,26,27,28,29},
  {0,4,6,8,10,11,12,13,14,15,16,17,21,27,29, 0,0,0,0},
  {0,4,6,10,12,15,21,24,25, 0,0,0,0,0,0,0,0,0,0}
};

// REMAP[h][label] = index of label in c_cand[h], else 0
__constant__ int c_remap[NH][NLAB] = {
  {0,1,2,3,0,4,0,5,0,0,0,0,0,0,0,0,0,0,0,6,7,0,0,0,0,0,0,0,8,0},
  {0,1,2,3,0,4,0,5,0,0,0,0,0,0,0,0,0,0,0,6,7,0,0,0,0,0,0,0,8,0},
  {0,0,0,0,0,1,0,2,0,0,0,0,0,0,0,0,0,0,3,4,5,0,6,0,0,0,0,0,0,0},
  {0,1,2,3,0,4,0,5,0,0,0,0,0,0,0,0,0,0,0,6,7,0,0,0,0,0,0,0,0,0},
  {0,1,2,3,0,4,0,5,0,0,0,0,0,0,0,0,0,0,0,6,7,0,0,0,0,0,0,0,8,0},
  {0,1,2,3,0,4,0,5,0,6,0,0,0,0,0,0,0,0,0,0,7,0,0,0,0,0,0,0,0,0},
  {0,0,1,0,2,0,3,0,4,0,5,0,6,7,8,9,10,11,0,0,0,12,0,0,13,0,14,15,0,0},
  {0,0,0,0,1,0,2,0,3,0,4,5,6,7,8,9,0,10,0,0,0,0,0,11,12,0,13,14,0,15},
  {0,0,0,0,1,0,2,0,0,0,3,4,0,0,5,6,0,7,0,0,0,8,0,0,9,10,11,12,0,0},
  {0,0,0,0,1,0,2,3,4,0,5,6,7,8,9,10,0,11,0,0,0,12,0,13,14,0,15,16,17,18},
  {0,0,0,0,1,0,2,0,3,0,4,5,6,7,8,9,10,11,0,0,0,12,0,0,0,0,0,13,0,14},
  {0,0,0,0,1,0,2,0,0,0,3,0,4,0,0,5,0,0,0,0,0,6,0,0,7,8,0,0,0,0}
};

// Branch-free tanh: hardware v_tanh_f32 if available, else exp/rcp sequence.
__device__ __forceinline__ float fast_tanh(float x) {
#if __has_builtin(__builtin_amdgcn_tanhf)
  return __builtin_amdgcn_tanhf(x);
#else
  float e = __expf(2.0f * x);                      // v_exp_f32 path
  return 1.0f - 2.0f * __builtin_amdgcn_rcpf(e + 1.0f);
#endif
}

// ---------------------------------------------------------------------------
// Pack Ws[12][1024][19] f32 -> Bt[144][1024] f16 (K-contiguous per packed col)
// ---------------------------------------------------------------------------
__global__ void pack_weights(const float* __restrict__ Ws,
                             _Float16* __restrict__ Bt) {
  int idx = blockIdx.x * blockDim.x + threadIdx.x;
  if (idx >= NCOLP * HID) return;
  int col = idx / HID;
  int k   = idx % HID;
  float v = 0.0f;
  if (col < NCOL) {
    int h = 0;
#pragma unroll
    for (int i = 1; i < NH; ++i) if (col >= c_off[i]) h = i;
    int j = col - c_off[h];
    v = Ws[((size_t)h * HID + k) * MAXO + j];
  }
  Bt[(size_t)col * HID + k] = (_Float16)v;
}

__global__ void zero_accum(float* __restrict__ a) {
  if (threadIdx.x < 32) a[threadIdx.x] = 0.0f;
}

__device__ __forceinline__ void load_a4(v4f a[4], const float* __restrict__ ap,
                                        int kc) {
  a[0] = __builtin_nontemporal_load((const v4f*)(ap + kc));
  a[1] = __builtin_nontemporal_load((const v4f*)(ap + kc + 4));
  a[2] = __builtin_nontemporal_load((const v4f*)(ap + kc + 16));
  a[3] = __builtin_nontemporal_load((const v4f*)(ap + kc + 20));
}

__device__ __forceinline__ v16h cvt_tanh(const v4f a[4]) {
  v16h af;
#pragma unroll
  for (int j = 0; j < 4; ++j) {
    af[j]      = (_Float16)fast_tanh(a[0][j]);
    af[4 + j]  = (_Float16)fast_tanh(a[1][j]);
    af[8 + j]  = (_Float16)fast_tanh(a[2][j]);
    af[12 + j] = (_Float16)fast_tanh(a[3][j]);
  }
  return af;
}

// ---------------------------------------------------------------------------
// Fused tanh + multi-head GEMM: C[B][144] = tanh(features) @ Bt^T
// 256 threads = 8 waves; wave w owns rows [blk*128 + w*16, +16), all 9 N-tiles.
// B slices staged in LDS (shared by 8 waves), A double-buffered in registers.
// ---------------------------------------------------------------------------
__global__ __launch_bounds__(256) void gemm_tanh_wmma(
    const float* __restrict__ features,
    const _Float16* __restrict__ Bt,
    float* __restrict__ C) {
  __shared__ _Float16 sB[NCOLP * SBS];          // 144*136*2 = 39,168 B

  const int tid  = threadIdx.x;
  const int wave = tid >> 5;
  const int lane = tid & 31;
  const int hb   = lane >> 4;                   // lane half: K sub-block select
  const int l16  = lane & 15;
  const int m0   = blockIdx.x * 128 + wave * 16;

  // A fragment (ISA 16-bit A 16x32): lane<16 K=base+0..7 & +16..23 of row l16;
  // lane>=16 K=base+8..15 & +24..31.
  const float*    ap  = features + (size_t)(m0 + l16) * HID + hb * 8;
  // B fragment (ISA 16-bit B 32x16): lane n holds col n; lane<16 K 0..7 & 16..23, lane>=16 K 8..15 & 24..31.
  const _Float16* sbp = sB + l16 * SBS + hb * 8;

  v8f acc[NT] = {};
  v4f aa[2][4];
  load_a4(aa[0], ap, 0);                        // preload first A chunk

  for (int kg = 0; kg < HID / KG; ++kg) {
    if (kg) __syncthreads();                    // previous slice fully consumed
    // Cooperative stage of B slice kg: 144 cols x 128 K halves (36 KB).
#pragma unroll
    for (int i = 0; i < 9; ++i) {
      int s = i * 256 + tid;                    // 2304 x 16B segments
      int col = s >> 4, part = s & 15;
      *(v8h*)(sB + col * SBS + part * 8) =
          *(const v8h*)(Bt + (size_t)col * HID + kg * KG + part * 8);
    }
    __syncthreads();

#pragma unroll
    for (int cch = 0; cch < 4; ++cch) {         // four 32-K chunks per slice
      const int kc = kg * KG + cch * 32;
      const int ka = (kc + 32 < HID) ? kc + 32 : kc;
      load_a4(aa[(cch + 1) & 1], ap, ka);       // prefetch next A chunk
      v16h af = cvt_tanh(aa[cch & 1]);
#pragma unroll
      for (int t = 0; t < NT; ++t) {
        const _Float16* p = sbp + t * 16 * SBS + cch * 32;
        v8h b0 = *(const v8h*)(p);
        v8h b1 = *(const v8h*)(p + 16);
        v16h bf = __builtin_shufflevector(b0, b1, 0, 1, 2, 3, 4, 5, 6, 7,
                                          8, 9, 10, 11, 12, 13, 14, 15);
        acc[t] = __builtin_amdgcn_wmma_f32_16x16x32_f16(
            false, af, false, bf, (short)0, acc[t], false, false);
      }
    }
  }

  // C layout: VGPR r -> row m0 + hb*8 + r, col t*16 + l16
  float* crow = C + (size_t)(m0 + hb * 8) * NCOLP + l16;
#pragma unroll
  for (int t = 0; t < NT; ++t) {
#pragma unroll
    for (int r = 0; r < 8; ++r) {
      crow[(size_t)r * NCOLP + t * 16] = acc[t][r];
    }
  }
}

// ---------------------------------------------------------------------------
// Epilogue: per-sample softmax-CE for all 12 heads + scatter selected head's
// logits into 30-label space.  Wave-level reduction before loss atomics.
// ---------------------------------------------------------------------------
__global__ __launch_bounds__(256) void epilogue(
    const float* __restrict__ C, const float* __restrict__ bs,
    const int* __restrict__ head_ids, const int* __restrict__ labels,
    float* __restrict__ out, float* __restrict__ accum) {
  const int b   = blockIdx.x * blockDim.x + threadIdx.x;
  const int lab = labels[b];
  const int sel = head_ids[b];
  const float* row = C + (size_t)b * NCOLP;

  float full[NLAB];
#pragma unroll
  for (int i = 0; i < NLAB; ++i) full[i] = 1e-7f;

  float num[NH], den[NH];
  for (int h = 0; h < NH; ++h) {
    const int n = c_n[h], off = c_off[h];
    float lg[MAXO];
    float m = -3.4e38f;
    for (int j = 0; j < n; ++j) {
      lg[j] = row[off + j] + bs[h * MAXO + j];
      m = fmaxf(m, lg[j]);
    }
    float s = 0.0f;
    for (int j = 0; j < n; ++j) s += __expf(lg[j] - m);
    int y = c_remap[h][lab];
    float nll = -(lg[y] - m - __logf(s));
    float w = (y == 0) ? 0.05f : 1.0f;
    num[h] = w * nll;
    den[h] = w;
    if (h == sel) {
      for (int j = 0; j < n; ++j) full[c_cand[h][j]] = lg[j];
    }
  }

  float* op = out + 1 + (size_t)b * NLAB;
  for (int i = 0; i < NLAB; ++i) op[i] = full[i];

  // reduce (num, den) across the wave32, then one atomic per value per wave
#pragma unroll
  for (int h = 0; h < NH; ++h) {
    float vn = num[h], vd = den[h];
    for (int o = 16; o > 0; o >>= 1) {
      vn += __shfl_down(vn, o, 32);
      vd += __shfl_down(vd, o, 32);
    }
    if ((threadIdx.x & 31) == 0) {
      atomicAdd(&accum[h], vn);
      atomicAdd(&accum[NH + h], vd);
    }
  }
}

__global__ void finalize(const float* __restrict__ accum,
                         float* __restrict__ out) {
  if (threadIdx.x == 0) {
    float s = 0.0f;
    for (int h = 0; h < NH; ++h) s += accum[h] / accum[NH + h];
    out[0] = s;
  }
}

// ---------------------------------------------------------------------------
extern "C" void kernel_launch(void* const* d_in, const int* in_sizes, int n_in,
                              void* d_out, int out_size, void* d_ws, size_t ws_size,
                              hipStream_t stream) {
  const float* features = (const float*)d_in[0];  // [65536,1024]
  const float* Ws       = (const float*)d_in[1];  // [12,1024,19]
  const float* bs       = (const float*)d_in[2];  // [12,19]
  const int*   head_ids = (const int*)d_in[3];    // [65536]
  const int*   labels   = (const int*)d_in[4];    // [65536]
  float* out = (float*)d_out;                     // [1 + 65536*30]

  char* ws = (char*)d_ws;
  _Float16* Bt  = (_Float16*)ws;                  // 144*1024*2 = 294,912 B
  float* accum  = (float*)(ws + 294912);          // 32 floats
  float* C      = (float*)(ws + 327680);          // 65536*144*4 = 37.75 MB

  pack_weights<<<(NCOLP * HID + 255) / 256, 256, 0, stream>>>(Ws, Bt);
  zero_accum<<<1, 32, 0, stream>>>(accum);
  gemm_tanh_wmma<<<BATCH / 128, 256, 0, stream>>>(features, Bt, C);
  epilogue<<<BATCH / 256, 256, 0, stream>>>(C, bs, head_ids, labels, out, accum);
  finalize<<<1, 1, 0, stream>>>(accum, out);
}
